// CrossAttention_19696720019990
// MI455X (gfx1250) — matrix operand order repaired and checked by reading, not compiled, not run on hardware
//
#include <hip/hip_runtime.h>

// ---------------------------------------------------------------------------
// CDNA5 (gfx1250) fused cross-attention, round 4:
//   - bf16 WMMA everywhere; K-contiguous operand layouts -> all fragments are
//     b128 pairs
//   - proj GEMM is LDS-free (direct-global fragments, clause-pipelined)
//   - flash attention: k-tile staged into LDS with GLOBAL_LOAD_ASYNC_TO_LDS_B128
//     (ASYNCcnt-tracked, VGPR-bypassing), v fragments direct global,
//     P relayout through wave-private LDS, vectorized epilogue
// ---------------------------------------------------------------------------

typedef __attribute__((ext_vector_type(16))) __bf16       v16bf;
typedef __attribute__((ext_vector_type(8)))  float        v8f;
typedef __attribute__((ext_vector_type(4)))  unsigned int v4u;
typedef __attribute__((ext_vector_type(4)))  float        v4f;

#define BATCH 16
#define CHN   256   // C = K = V = 256
#define NPIX  1024  // H*W
#define TDIM  512

union FragA { v16bf v; v4u q[2]; };
union FragB { v16bf v; v4u q[2]; };
union Pack8 { __bf16 h[8]; v4u q; };

// A fragment (16x32): lane (half,m) = runs K[half*16..+15].
__device__ inline v16bf load_frag_a2(const __bf16* base, int lda, int lane) {
  int half = lane >> 4, m = lane & 15;
  FragA f;
  const __bf16* row = base + m * lda + half * 16;
  f.q[0] = *(const v4u*)(row);
  f.q[1] = *(const v4u*)(row + 8);
  return f.v;
}

// B fragment (32x16) from K-contiguous [N][K] storage: lane (half,n) reads
// B[half*16 .. +15][n] = 32 contiguous bytes.
__device__ inline v16bf load_frag_bT(const __bf16* baseT, int ldk, int lane) {
  int half = lane >> 4, n = lane & 15;
  FragB f;
  const __bf16* p = baseT + n * ldk + half * 16;
  f.q[0] = *(const v4u*)(p);
  f.q[1] = *(const v4u*)(p + 8);
  return f.v;
}

// ---------------------------------------------------------------------------
// f32 -> bf16 convert (weights): 8 elts/thread.
// ---------------------------------------------------------------------------
__global__ __launch_bounds__(256)
void cvt_bf16_kernel(const float* __restrict__ src, __bf16* __restrict__ dst) {
  int i = (blockIdx.x * 256 + threadIdx.x) * 8;
  v4f f0 = *(const v4f*)(src + i);
  v4f f1 = *(const v4f*)(src + i + 4);
  Pack8 pk;
#pragma unroll
  for (int j = 0; j < 4; ++j) { pk.h[j] = (__bf16)f0[j]; pk.h[4 + j] = (__bf16)f1[j]; }
  *(v4u*)(dst + i) = pk.q;
}

// ---------------------------------------------------------------------------
// f32 [B][C][N] -> bf16 [B][N][C] tiled transpose+convert (64x64 tiles).
// ---------------------------------------------------------------------------
__global__ __launch_bounds__(256)
void transpose_cvt_kernel(const float* __restrict__ src, __bf16* __restrict__ dst) {
  __shared__ __align__(16) __bf16 t[64 * 72];   // [n][c], pad 64->72 (144B rows)
  int b = blockIdx.z, c0 = blockIdx.y * 64, n0 = blockIdx.x * 64;
  int tid = threadIdx.x;
  {
    int e = tid * 16;               // 4096 elts, 16/thread
    int cc = e >> 6, nn = e & 63;   // row c, 16 consecutive n
    const float* gp = src + ((size_t)b * CHN + c0 + cc) * NPIX + n0 + nn;
#pragma unroll
    for (int j = 0; j < 16; j += 4) {
      v4f f = *(const v4f*)(gp + j);
#pragma unroll
      for (int u = 0; u < 4; ++u) t[(nn + j + u) * 72 + cc] = (__bf16)f[u];
    }
  }
  __syncthreads();
  {
    int e = tid * 16;
    int nn = e >> 6, cc = e & 63;   // row n, 16 consecutive c
    __bf16* gp = dst + ((size_t)b * NPIX + n0 + nn) * CHN + c0 + cc;
    const __bf16* lp = t + nn * 72 + cc;
    *(v4u*)(gp)     = *(const v4u*)(lp);
    *(v4u*)(gp + 8) = *(const v4u*)(lp + 8);
  }
}

// ---------------------------------------------------------------------------
// LDS-free projection GEMM: Out[b,o,n] = sum_i Wb[o,i]*InT[b,n,i] + bias[o].
// Wb bf16 [O][I]; InT bf16 [B][N][I]. out_nmajor=1 -> Out [b][n][o].
// ---------------------------------------------------------------------------
__global__ __launch_bounds__(256)
void proj_kernel(const __bf16* __restrict__ Wb, const float* __restrict__ bias,
                 const __bf16* __restrict__ InT, __bf16* __restrict__ Out,
                 int O, int I, int N, int out_nmajor) {
  int b = blockIdx.z, o0 = blockIdx.y * 64, n0 = blockIdx.x * 64;
  int lane = threadIdx.x & 31, wave = threadIdx.x >> 5;
  int osub = wave >> 1, nsub0 = (wave & 1) * 2;
  int half = lane >> 4, lcol = lane & 15;
  v8f acc0 = {}; v8f acc1 = {};

  const __bf16* arow  = Wb + (size_t)(o0 + osub * 16 + lcol) * I + half * 16;
  const __bf16* b0row = InT + ((size_t)b * N + n0 + nsub0 * 16 + lcol) * I + half * 16;
  const __bf16* b1row = b0row + (size_t)16 * I;

#pragma unroll
  for (int i0 = 0; i0 < 256; i0 += 32) {
    FragA a;  a.q[0]  = *(const v4u*)(arow + i0);  a.q[1]  = *(const v4u*)(arow + i0 + 8);
    FragB f0; f0.q[0] = *(const v4u*)(b0row + i0); f0.q[1] = *(const v4u*)(b0row + i0 + 8);
    FragB f1; f1.q[0] = *(const v4u*)(b1row + i0); f1.q[1] = *(const v4u*)(b1row + i0 + 8);
    acc0 = __builtin_amdgcn_wmma_f32_16x16x32_bf16(false, a.v, false, f0.v, (short)0, acc0, false, false);
    acc1 = __builtin_amdgcn_wmma_f32_16x16x32_bf16(false, a.v, false, f1.v, (short)0, acc1, false, false);
  }

  int obase = o0 + osub * 16 + half * 8;   // 8 consecutive o per accumulator
  int n0g = n0 + nsub0 * 16 + lcol;
  if (out_nmajor) {
    Pack8 p0, p1;
#pragma unroll
    for (int r = 0; r < 8; ++r) {
      float bia = bias[obase + r];
      p0.h[r] = (__bf16)(acc0[r] + bia);
      p1.h[r] = (__bf16)(acc1[r] + bia);
    }
    *(v4u*)(Out + (size_t)b * N * O + (size_t)n0g * O + obase)        = p0.q;
    *(v4u*)(Out + (size_t)b * N * O + (size_t)(n0g + 16) * O + obase) = p1.q;
  } else {
#pragma unroll
    for (int r = 0; r < 8; ++r) {
      float bia = bias[obase + r];
      Out[(size_t)b * O * N + (size_t)(obase + r) * N + n0g]      = (__bf16)(acc0[r] + bia);
      Out[(size_t)b * O * N + (size_t)(obase + r) * N + n0g + 16] = (__bf16)(acc1[r] + bia);
    }
  }
}

// ---------------------------------------------------------------------------
// temb projection: tout[b,c] = bt[c] + sum_t swish(temb[b,t]) * Wt[c,t]
// ---------------------------------------------------------------------------
__global__ __launch_bounds__(256)
void temb_proj_kernel(const float* __restrict__ temb, const float* __restrict__ Wt,
                      const float* __restrict__ bt, float* __restrict__ tout) {
  __shared__ float sw[TDIM];
  int b = blockIdx.x, tid = threadIdx.x;
  for (int t = tid; t < TDIM; t += 256) {
    float v = temb[b * TDIM + t];
    sw[t] = v / (1.f + __expf(-v));
  }
  __syncthreads();
  int c = tid;
  const float* wr = Wt + (size_t)c * TDIM;
  float s = bt[c];
  for (int t = 0; t < TDIM; ++t) s += sw[t] * wr[t];
  tout[b * CHN + c] = s;
}

// ---------------------------------------------------------------------------
// Flash attention + fused epilogue.
// Block: (batch b, 64-row n-tile). 8 waves: (nsub = w>>1, chalf = w&1).
// S[n,m] = sum_k q[k,n]*k[k,m]; online softmax over m; O[n,c] += P[n,m]*v[c,m].
// out[b,c,n] = gamma * O/rowsum + x[b,c,n] + tproj[b,c].
// k tiles staged via GLOBAL_LOAD_ASYNC_TO_LDS_B128 (ASYNCcnt-tracked).
// ---------------------------------------------------------------------------
__global__ __launch_bounds__(256)
void attn_kernel(const __bf16* __restrict__ qT,   // [B][N][256]
                 const __bf16* __restrict__ kT,   // [B][N][256]
                 const __bf16* __restrict__ vm,   // [B][256][N]
                 const float*  __restrict__ x,    // [B][256][N]
                 const float*  __restrict__ tproj,// [B][256]
                 const float*  __restrict__ gamma,
                 float* __restrict__ out) {
  __shared__ __align__(16) __bf16 kTs[64 * 72];     // [m][kk-chunk], pad 64->72
  __shared__ __align__(16) __bf16 ps[8 * 16 * 72];  // per-wave P [16 n][64 m]
  int b = blockIdx.y, n0 = blockIdx.x * 64;
  int tid = threadIdx.x, lane = tid & 31, wave = tid >> 5;
  int half = lane >> 4, lcol = lane & 15;
  int nsub = wave >> 1, chalf = wave & 1;
  __bf16* pw = ps + wave * (16 * 72);

  // Preload all 8 q^T A-fragments with b128 global loads.
  FragA qf[8];
  {
    const __bf16* qrow = qT + ((size_t)b * NPIX + (n0 + nsub * 16 + lcol)) * 256 + half * 16;
#pragma unroll
    for (int ksv = 0; ksv < 8; ++ksv) {
      qf[ksv].q[0] = *(const v4u*)(qrow + ksv * 32);
      qf[ksv].q[1] = *(const v4u*)(qrow + ksv * 32 + 8);
    }
  }
  // Per-lane base row for direct-global v fragments: row c, m-contiguous.
  const __bf16* vbase = vm + ((size_t)b * CHN + chalf * 128 + lcol) * NPIX + half * 16;

  // Per-thread staging addresses for the async k-tile copies.
  int se = tid * 16;
  int srow = se >> 6, scol = se & 63;
  const __bf16* kgbase = kT + ((size_t)b * NPIX + srow) * 256 + scol;  // + m0*256 + kst*64
  unsigned klds = (unsigned)(size_t)(kTs + srow * 72 + scol);  // LDS offset in addr[31:0]

  v8f acc[8];
#pragma unroll
  for (int i = 0; i < 8; ++i) acc[i] = (v8f){};
  float rmax[8], rsum[8];
#pragma unroll
  for (int r = 0; r < 8; ++r) { rmax[r] = -1e30f; rsum[r] = 0.f; }

  for (int mt = 0; mt < 16; ++mt) {
    int m0 = mt * 64;
    v8f s[4];
#pragma unroll
    for (int i = 0; i < 4; ++i) s[i] = (v8f){};

    // ---- S = q^T x k over 256 channels (4 async-staged 64-k chunks) ----
    for (int kst = 0; kst < 4; ++kst) {
      __syncthreads();   // previous chunk fully consumed
      {
        const __bf16* gp = kgbase + (size_t)m0 * 256 + kst * 64;
        unsigned long long gaddr = (unsigned long long)(size_t)gp;
        // Async DMA: global -> LDS, 2x16B per lane, bypasses VGPRs (ASYNCcnt).
        asm volatile(
            "global_load_async_to_lds_b128 %0, %1, off\n\t"
            "global_load_async_to_lds_b128 %0, %1, off offset:16\n\t"
            "s_wait_asynccnt 0x0"
            :
            : "v"(klds), "v"(gaddr)
            : "memory");
        __builtin_prefetch(gp + 64 * 256, 0, 1);   // next m-tile (global_prefetch)
      }
      __syncthreads();   // chunk visible to all waves
#pragma unroll
      for (int k2 = 0; k2 < 2; ++k2) {
        v16bf a = qf[kst * 2 + k2].v;
#pragma unroll
        for (int ms = 0; ms < 4; ++ms) {
          v16bf bb = load_frag_bT(kTs + ms * 16 * 72 + k2 * 32, 72, lane);
          s[ms] = __builtin_amdgcn_wmma_f32_16x16x32_bf16(false, a, false, bb, (short)0, s[ms], false, false);
        }
      }
    }

    // ---- online softmax; D-layout row = r + half*8, intra-half reductions ----
#pragma unroll
    for (int r = 0; r < 8; ++r) {
      float lm = s[0][r];
#pragma unroll
      for (int ms = 1; ms < 4; ++ms) lm = fmaxf(lm, s[ms][r]);
#pragma unroll
      for (int off = 1; off < 16; off <<= 1)
        lm = fmaxf(lm, __shfl_xor(lm, off, 32));
      float nm = fmaxf(rmax[r], lm);
      float scale = __expf(rmax[r] - nm);
      rmax[r] = nm;
      float ls = 0.f;
#pragma unroll
      for (int ms = 0; ms < 4; ++ms) {
        float p = __expf(s[ms][r] - nm);
        ls += p;
        pw[(r + half * 8) * 72 + ms * 16 + lcol] = (__bf16)p;  // C/D -> A relayout
      }
#pragma unroll
      for (int off = 1; off < 16; off <<= 1)
        ls += __shfl_xor(ls, off, 32);
      rsum[r] = rsum[r] * scale + ls;
#pragma unroll
      for (int ct = 0; ct < 8; ++ct) acc[ct][r] *= scale;
    }

    // ---- O += P x v^T: A from wave-private LDS, B direct global ----
#pragma unroll
    for (int vst = 0; vst < 2; ++vst) {
      v16bf ap = load_frag_a2(pw + vst * 32, 72, lane);
      const __bf16* vrow = vbase + m0 + vst * 32;
#pragma unroll
      for (int ct = 0; ct < 8; ++ct) {
        FragB bb;
        bb.q[0] = *(const v4u*)(vrow + (size_t)ct * 16 * NPIX);
        bb.q[1] = *(const v4u*)(vrow + (size_t)ct * 16 * NPIX + 8);
        acc[ct] = __builtin_amdgcn_wmma_f32_16x16x32_bf16(false, ap, false, bb.v, (short)0, acc[ct], false, false);
      }
    }
  }

  // ---- epilogue: out = gamma*softmax_out + x + tproj, vectorized in n ----
  float g = gamma[0];
  float inv[8];
#pragma unroll
  for (int r = 0; r < 8; ++r) inv[r] = 1.f / rsum[r];
  int nbase = n0 + nsub * 16 + half * 8;   // 8 consecutive n per accumulator
#pragma unroll
  for (int ct = 0; ct < 8; ++ct) {
    int c = chalf * 128 + ct * 16 + lcol;
    float tp = tproj[b * CHN + c];
    const float* xp = x   + ((size_t)b * CHN + c) * NPIX + nbase;
    float*       op = out + ((size_t)b * CHN + c) * NPIX + nbase;
    v4f x0 = *(const v4f*)(xp);
    v4f x1 = *(const v4f*)(xp + 4);
    v4f o0, o1;
#pragma unroll
    for (int r = 0; r < 4; ++r) {
      o0[r] = g * (acc[ct][r]     * inv[r])     + x0[r] + tp;
      o1[r] = g * (acc[ct][4 + r] * inv[4 + r]) + x1[r] + tp;
    }
    *(v4f*)(op)     = o0;
    *(v4f*)(op + 4) = o1;
  }
}

// ---------------------------------------------------------------------------
extern "C" void kernel_launch(void* const* d_in, const int* in_sizes, int n_in,
                              void* d_out, int out_size, void* d_ws, size_t ws_size,
                              hipStream_t stream) {
  const float* x        = (const float*)d_in[0];
  const float* key_in   = (const float*)d_in[1];
  const float* value_in = (const float*)d_in[2];
  const float* temb     = (const float*)d_in[3];
  const float* Wq       = (const float*)d_in[4];
  const float* bq       = (const float*)d_in[5];
  const float* Wk       = (const float*)d_in[6];
  const float* bk       = (const float*)d_in[7];
  const float* Wv       = (const float*)d_in[8];
  const float* bv       = (const float*)d_in[9];
  const float* gamma    = (const float*)d_in[10];
  const float* Wt       = (const float*)d_in[11];
  const float* bt       = (const float*)d_in[12];

  char* ws = (char*)d_ws;
  const size_t MB = (size_t)1 << 20;
  __bf16* xT   = (__bf16*)(ws);             // [16][1024][256] bf16, 8 MB
  __bf16* keyT = (__bf16*)(ws + 8  * MB);   // [16][1024][256]
  __bf16* valT = (__bf16*)(ws + 16 * MB);   // [16][1024][256]
  __bf16* qT   = (__bf16*)(ws + 24 * MB);   // [16][1024][256]
  __bf16* kTw  = (__bf16*)(ws + 32 * MB);   // [16][1024][256]
  __bf16* vmw  = (__bf16*)(ws + 40 * MB);   // [16][256][1024]
  __bf16* Wqb  = (__bf16*)(ws + 48 * MB);
  __bf16* Wkb  = (__bf16*)(ws + 48 * MB + 128 * 1024);
  __bf16* Wvb  = (__bf16*)(ws + 48 * MB + 256 * 1024);
  float*  tpw  = (float*) (ws + 48 * MB + 512 * 1024);

  // Pre-passes: weight converts + input transpose/converts.
  cvt_bf16_kernel<<<dim3(32), dim3(256), 0, stream>>>(Wq, Wqb);
  cvt_bf16_kernel<<<dim3(32), dim3(256), 0, stream>>>(Wk, Wkb);
  cvt_bf16_kernel<<<dim3(32), dim3(256), 0, stream>>>(Wv, Wvb);
  dim3 tg(16, 4, 16);
  transpose_cvt_kernel<<<tg, dim3(256), 0, stream>>>(x,        xT);
  transpose_cvt_kernel<<<tg, dim3(256), 0, stream>>>(key_in,   keyT);
  transpose_cvt_kernel<<<tg, dim3(256), 0, stream>>>(value_in, valT);

  // Projections (LDS-free WMMA GEMMs).
  dim3 pg(16, 4, 16);   // (N/64, O/64, B)
  proj_kernel<<<pg, dim3(256), 0, stream>>>(Wqb, bq, xT,   qT,  256, 256, 1024, 1);
  proj_kernel<<<pg, dim3(256), 0, stream>>>(Wkb, bk, keyT, kTw, 256, 256, 1024, 1);
  proj_kernel<<<pg, dim3(256), 0, stream>>>(Wvb, bv, valT, vmw, 256, 256, 1024, 0);

  temb_proj_kernel<<<dim3(16), dim3(256), 0, stream>>>(temb, Wt, bt, tpw);
  attn_kernel<<<dim3(16, 16), dim3(256), 0, stream>>>(qT, kTw, vmw, x, tpw, gamma,
                                                      (float*)d_out);
}
